// MultimodalFusion_36790689857881
// MI455X (gfx1250) — compile-verified, matrix operand order checked by reading, not run on hardware
//
#include <hip/hip_runtime.h>
#include <hip/hip_bf16.h>

// ---------------------------------------------------------------------------
// MultimodalFusion (KAN tabular encoder + 2x Mamba + cross-attn) for gfx1250
// GEMMs: v_wmma_f32_16x16x32_f16, weights staged to LDS via the Tensor Data
// Mover (tensor_load_to_lds + s_wait_tensorcnt), A fragments software-
// pipelined from global with f32->f16 convert in-register.
// ---------------------------------------------------------------------------

typedef _Float16 v16h  __attribute__((ext_vector_type(16)));
typedef _Float16 h8    __attribute__((ext_vector_type(8)));
typedef float    v8f   __attribute__((ext_vector_type(8)));
typedef float    f32x4 __attribute__((ext_vector_type(4)));
typedef unsigned int u32x4 __attribute__((ext_vector_type(4)));
typedef int      i32x8 __attribute__((ext_vector_type(8)));
typedef int      i32x4 __attribute__((ext_vector_type(4)));

#define BATCH 32
#define SEQL  2048
#define TEMPD 32
#define HD    128
#define H2D   64
#define TABD  64
#define DI    256
#define DS    16
#define DR    8
#define NXD   40   // DR + 2*DS
#define BL    ((long)BATCH * SEQL)

#if __has_builtin(__builtin_amdgcn_tensor_load_to_lds) && \
    __has_builtin(__builtin_amdgcn_s_wait_tensorcnt)
#define HAS_TDM 1
#else
#define HAS_TDM 0
#endif

// ------------------------- fragment helpers --------------------------------
// A fragment (16x32 f16, ISA 7.12.2): lane l holds row (l&15); elements
// 0..7 -> K = kb+kbase+0..7, elements 8..15 -> K = kb+kbase+16..23,
// kbase = (l>=16)*8.  Same pattern serves B fragments from W rows (W: NxK).

__device__ __forceinline__ v16h load_a_frag(const float* __restrict__ p) {
    f32x4 q0 = *(const f32x4*)(p);
    f32x4 q1 = *(const f32x4*)(p + 4);
    f32x4 q2 = *(const f32x4*)(p + 16);
    f32x4 q3 = *(const f32x4*)(p + 20);
    v16h f;
#pragma unroll
    for (int e = 0; e < 4; ++e) {
        f[e]      = (_Float16)q0[e];
        f[4 + e]  = (_Float16)q1[e];
        f[8 + e]  = (_Float16)q2[e];
        f[12 + e] = (_Float16)q3[e];
    }
    return f;
}

__device__ __forceinline__ v16h load_b_frag(const _Float16* p) {
    h8 lo = *(const h8*)(p);
    h8 hi = *(const h8*)(p + 16);
    v16h f;
#pragma unroll
    for (int e = 0; e < 8; ++e) { f[e] = lo[e]; f[8 + e] = hi[e]; }
    return f;
}

#if HAS_TDM
// TDM: copy nd8 8-byte elements (one 1-D tile) from global f16 weights to LDS.
// D# packing per CDNA5 ISA (group0/group1).
// 6-arg builtin form: (g0 u32x4, g1 i32x8, g2 i32x4, g3 i32x4, g4 i32x8, cpol)
__device__ __forceinline__ void tdm_load_tile(const _Float16* gsrc, _Float16* ldst,
                                              unsigned int nd8) {
    unsigned long long ga = (unsigned long long)(size_t)gsrc;
    unsigned int la = (unsigned int)(size_t)(void*)ldst;   // LDS byte offset
    u32x4 g0;
    g0[0] = 1u;                                            // count=1, user mode
    g0[1] = la;                                            // lds_addr
    g0[2] = (unsigned int)(ga & 0xFFFFFFFFu);              // global_addr[31:0]
    g0[3] = (unsigned int)((ga >> 32) & 0x1FFFFFFu)        // global_addr[56:32]
          | (2u << 30);                                    // type = 2 (image)
    i32x8 g1;
    g1[0] = (int)(3u << 16);                               // data_size = 8B
    g1[1] = (int)((nd8 & 0xFFFFu) << 16);                  // tensor_dim0[15:0]
    g1[2] = (int)((nd8 >> 16) | (1u << 16));               // tensor_dim0[31:16], tensor_dim1=1
    g1[3] = (int)((nd8 & 0xFFFFu) << 16);                  // tile_dim0 = nd8
    g1[4] = 1;                                             // tile_dim1 = 1
    g1[5] = (int)nd8;                                      // tensor_dim0_stride
    g1[6] = 0;
    g1[7] = 0;
    i32x4 g2 = {0, 0, 0, 0};
    i32x4 g3 = {0, 0, 0, 0};
    i32x8 g4 = {0, 0, 0, 0, 0, 0, 0, 0};
    __builtin_amdgcn_tensor_load_to_lds(g0, g1, g2, g3, g4, 0);
}
#endif

// ------------------------- weight f32 -> f16 (padded) ----------------------
__global__ __launch_bounds__(256)
void wcvt_kernel(const float* __restrict__ W, _Float16* __restrict__ Wh,
                 int N, int K, long total) {
    long i = (long)blockIdx.x * blockDim.x + threadIdx.x;  // Npad*K
    if (i >= total) return;
    int n = (int)(i / K);
    Wh[i] = (n < N) ? (_Float16)W[i] : (_Float16)0.f;
}

// ------------------------- WMMA GEMM:  C = A @ W^T + bias ------------------
// A: MxK f32, Wh: (ngroups*64)xK f16 zero-padded, C: MxN f32.
// Grid: (M/128) * ngroups blocks; 8 waves/block, all sharing one n-group.
template<int K>
__global__ __launch_bounds__(256)
void gemm_wmma_kernel(const float* __restrict__ A, const _Float16* __restrict__ Wh,
                      const float* __restrict__ bias, float* __restrict__ C,
                      int M, int N) {
    __shared__ _Float16 wtile[64 * 256];    // up to 64 rows x K=256 (32 KB)

    const int m_blocks = M >> 7;            // 8 m-tiles of 16 rows per block
    const int mbg = blockIdx.x % m_blocks;
    const int ng  = blockIdx.x / m_blocks;

    // ---- stage this block's 64 x K weight tile into LDS -------------------
#if HAS_TDM
    if (threadIdx.x < 32) {                 // wave 0 issues the TDM DMA once
        tdm_load_tile(Wh + (long)ng * 64 * K, wtile, (unsigned)(16 * K));
        __builtin_amdgcn_s_wait_tensorcnt(0);
    }
    __syncthreads();
#else
    {
        const unsigned long long* src =
            (const unsigned long long*)(Wh + (long)ng * 64 * K);
        unsigned long long* dst = (unsigned long long*)wtile;
        for (int i = threadIdx.x; i < 16 * K; i += 256) dst[i] = src[i];
        __syncthreads();
    }
#endif

    const int lane  = threadIdx.x & 31;
    const int w     = threadIdx.x >> 5;
    const int mt    = mbg * 8 + w;
    const int lr    = lane & 15;
    const int half  = lane >> 4;
    const int kbase = half << 3;

    const float* arow = A + (long)(mt * 16 + lr) * K + kbase;
    const _Float16* brow[4];
#pragma unroll
    for (int t = 0; t < 4; ++t) brow[t] = wtile + (t * 16 + lr) * K + kbase;

    v8f acc[4] = {};
    v16h af = load_a_frag(arow);            // software-pipelined A fragment
#pragma unroll
    for (int kb = 0; kb < K; kb += 32) {
        v16h af_cur = af;
        if (kb + 32 < K) af = load_a_frag(arow + kb + 32);
#pragma unroll
        for (int t = 0; t < 4; ++t) {
            v16h bf = load_b_frag(brow[t] + kb);
            acc[t] = __builtin_amdgcn_wmma_f32_16x16x32_f16(
                false, af_cur, false, bf, (short)0, acc[t], false, false);
        }
    }

    // D layout: VGPR r -> row (mt*16 + r + half*8), col = lr  (ISA 7.12.2)
#pragma unroll
    for (int t = 0; t < 4; ++t) {
        int col = (ng * 4 + t) * 16 + lr;
        if (col < N) {
            float bv = bias ? bias[col] : 0.f;
#pragma unroll
            for (int r = 0; r < 8; ++r) {
                long mrow = (long)mt * 16 + r + (half << 3);
                C[mrow * N + col] = acc[t][r] + bv;
            }
        }
    }
}

// ------------------------- Tabular KAN encoder -----------------------------
__device__ __forceinline__ float kan_basis(float x, float grid) {
    float d = fabsf(x - grid);
    if (d < 1.f)  return 1.f + 3.f * d * d * (d - 1.f);    // 1 - 3d^2 + 3d^3
    if (d < 2.f) { float u = 2.f - d; return u * u * u; }
    return 0.f;
}

__global__ __launch_bounds__(128)
void kan_tab_kernel(const float* __restrict__ x,
                    const float* __restrict__ c1, const float* __restrict__ bw1,
                    const float* __restrict__ g1, const float* __restrict__ be1,
                    const float* __restrict__ c2, const float* __restrict__ bw2,
                    const float* __restrict__ g2, const float* __restrict__ be2,
                    const float* __restrict__ ow, const float* __restrict__ ob,
                    float* __restrict__ tab) {
    __shared__ float sin_[HD];
    __shared__ float smid[HD];
    __shared__ float red[HD];
    int b = blockIdx.x, t = threadIdx.x;

    if (t < TABD) sin_[t] = x[b * TABD + t];
    __syncthreads();

    float acc = 0.f;                                       // KAN1: 64 -> 128
    for (int i = 0; i < TABD; ++i) {
        float xv = sin_[i];
        acc += xv * bw1[t * TABD + i];
#pragma unroll
        for (int g = 0; g < 5; ++g)
            acc += kan_basis(xv, -1.f + 0.5f * (float)g) * c1[(t * TABD + i) * 5 + g];
    }
    red[t] = acc; __syncthreads();
    for (int s = 64; s > 0; s >>= 1) { if (t < s) red[t] += red[t + s]; __syncthreads(); }
    float mean = red[0] * (1.f / 128.f); __syncthreads();
    float dev = acc - mean;
    red[t] = dev * dev; __syncthreads();
    for (int s = 64; s > 0; s >>= 1) { if (t < s) red[t] += red[t + s]; __syncthreads(); }
    float var = red[0] * (1.f / 128.f); __syncthreads();
    smid[t] = dev * rsqrtf(var + 1e-5f) * g1[t] + be1[t];
    __syncthreads();

    float acc2 = 0.f;                                      // KAN2: 128 -> 64
    if (t < H2D) {
        for (int i = 0; i < HD; ++i) {
            float xv = smid[i];
            acc2 += xv * bw2[t * HD + i];
#pragma unroll
            for (int g = 0; g < 5; ++g)
                acc2 += kan_basis(xv, -1.f + 0.5f * (float)g) * c2[(t * HD + i) * 5 + g];
        }
    }
    red[t] = (t < H2D) ? acc2 : 0.f; __syncthreads();
    for (int s = 32; s > 0; s >>= 1) { if (t < s) red[t] += red[t + s]; __syncthreads(); }
    float mean2 = red[0] * (1.f / 64.f); __syncthreads();
    float dev2 = acc2 - mean2;
    red[t] = (t < H2D) ? dev2 * dev2 : 0.f; __syncthreads();
    for (int s = 32; s > 0; s >>= 1) { if (t < s) red[t] += red[t + s]; __syncthreads(); }
    float var2 = red[0] * (1.f / 64.f); __syncthreads();
    if (t < H2D) sin_[t] = dev2 * rsqrtf(var2 + 1e-5f) * g2[t] + be2[t];
    __syncthreads();

    float o = ob[t];                                       // kan_out: 64 -> 128
    for (int i = 0; i < H2D; ++i) o += sin_[i] * ow[t * H2D + i];
    tab[b * HD + t] = o;
}

// ------------------------- Mamba elementwise pieces ------------------------
__device__ __forceinline__ float silu(float v) { return v / (1.f + expf(-v)); }

__global__ __launch_bounds__(256)
void conv_silu_kernel(const float* __restrict__ xz, const float* __restrict__ cw,
                      const float* __restrict__ cb, float* __restrict__ xbc) {
    long i = (long)blockIdx.x * blockDim.x + threadIdx.x;  // BL*DI threads
    int c = (int)(i & (DI - 1));
    long bl = i >> 8;
    int l = (int)(bl & (SEQL - 1));
    long row0 = bl - l;
    float acc = cb[c];
#pragma unroll
    for (int j = 0; j < 4; ++j) {
        int ls = l - 3 + j;
        if (ls >= 0) acc += xz[(row0 + ls) * (2 * DI) + c] * cw[c * 4 + j];
    }
    xbc[i] = silu(acc);
}

__global__ __launch_bounds__(256)
void dtproj_kernel(const float* __restrict__ xdbl, const float* __restrict__ dw,
                   const float* __restrict__ db, float* __restrict__ dt) {
    long i = (long)blockIdx.x * blockDim.x + threadIdx.x;  // BL*DI threads
    int d = (int)(i & (DI - 1));
    long bl = i >> 8;
    float acc = db[d];
    const float* xr = xdbl + bl * NXD;
#pragma unroll
    for (int r = 0; r < DR; ++r) acc += xr[r] * dw[d * DR + r];
    dt[i] = (acc > 20.f) ? acc : log1pf(expf(acc));
}

// selective scan: one thread per (b, d); h[16] in registers; sequential in L
__global__ __launch_bounds__(256)
void scan_kernel(const float* __restrict__ dt, const float* __restrict__ xbc,
                 const float* __restrict__ xdbl, const float* __restrict__ Alog,
                 float* __restrict__ y) {
    int b = blockIdx.x;
    int d = threadIdx.x;
    float a[DS], h[DS];
#pragma unroll
    for (int s = 0; s < DS; ++s) { a[s] = -expf(Alog[d * DS + s]); h[s] = 0.f; }
    const long base = (long)b * SEQL;
    for (int l = 0; l < SEQL; ++l) {
        long bl = base + l;
        float dtv = dt[bl * DI + d];
        float xv  = xbc[bl * DI + d];
        const float* Bt = xdbl + bl * NXD + DR;
        const float* Ct = Bt + DS;
        float dx = dtv * xv;
        float acc = 0.f;
#pragma unroll
        for (int s = 0; s < DS; ++s) {
            float dA = expf(dtv * a[s]);
            h[s] = dA * h[s] + dx * Bt[s];
            acc += h[s] * Ct[s];
        }
        y[bl * DI + d] = acc;
    }
}

__global__ __launch_bounds__(256)
void yact_kernel(float* __restrict__ y, const float* __restrict__ xbc,
                 const float* __restrict__ xz, const float* __restrict__ Dv) {
    long i = (long)blockIdx.x * blockDim.x + threadIdx.x;  // BL*DI threads
    int d = (int)(i & (DI - 1));
    long bl = i >> 8;
    float z = xz[bl * (2 * DI) + DI + d];
    y[i] = (y[i] + xbc[i] * Dv[d]) * silu(z);
}

// ------------------------- Attention ---------------------------------------
__global__ __launch_bounds__(128)
void qproj_kernel(const float* __restrict__ tab, const float* __restrict__ wq,
                  const float* __restrict__ bq, float* __restrict__ q) {
    __shared__ float s[HD];
    int b = blockIdx.x, t = threadIdx.x;
    s[t] = tab[b * HD + t];
    __syncthreads();
    float acc = bq[t];
    for (int i = 0; i < HD; ++i) acc += s[i] * wq[t * HD + i];
    q[b * HD + t] = acc;
}

__global__ __launch_bounds__(256)
void attn_kernel(const float* __restrict__ q, const float* __restrict__ Kb,
                 const float* __restrict__ Vb, float* __restrict__ aout) {
    __shared__ float sc[SEQL];
    __shared__ float red[256];
    __shared__ float qv[32];
    int b = blockIdx.x, hh = blockIdx.y, t = threadIdx.x;
    if (t < 32) qv[t] = q[b * HD + hh * 32 + t];
    __syncthreads();

    const float scale = 0.17677669529663687f;   // 1/sqrt(32)
    float lmax = -3.4e38f;
    for (int j = 0; j < 8; ++j) {
        int k = t + j * 256;
        const float* kr = Kb + ((long)b * SEQL + k) * HD + hh * 32;
        float dot = 0.f;
#pragma unroll
        for (int dd = 0; dd < 32; ++dd) dot += qv[dd] * kr[dd];
        dot *= scale;
        sc[k] = dot;
        lmax = fmaxf(lmax, dot);
    }
    red[t] = lmax; __syncthreads();
    for (int s = 128; s > 0; s >>= 1) { if (t < s) red[t] = fmaxf(red[t], red[t + s]); __syncthreads(); }
    float m = red[0]; __syncthreads();

    float lsum = 0.f;
    for (int j = 0; j < 8; ++j) {
        int k = t + j * 256;
        float e = expf(sc[k] - m);
        sc[k] = e;
        lsum += e;
    }
    red[t] = lsum; __syncthreads();
    for (int s = 128; s > 0; s >>= 1) { if (t < s) red[t] += red[t + s]; __syncthreads(); }
    float denom = red[0]; __syncthreads();

    int dd = t & 31, g = t >> 5;
    float acc = 0.f;
    for (int k = g; k < SEQL; k += 8)
        acc += sc[k] * Vb[((long)b * SEQL + k) * HD + hh * 32 + dd];
    red[t] = acc; __syncthreads();
    if (t < 32) {
        float tot = 0.f;
#pragma unroll
        for (int gg = 0; gg < 8; ++gg) tot += red[gg * 32 + t];
        aout[b * HD + hh * 32 + t] = tot / denom;
    }
}

// out-projection + heads; d_out = [risk(32) | forecast(32) | diet(160) | fused(4096)]
__global__ __launch_bounds__(128)
void head_kernel(const float* __restrict__ aout,
                 const float* __restrict__ ow, const float* __restrict__ ob,
                 const float* __restrict__ rw, const float* __restrict__ rb,
                 const float* __restrict__ fw, const float* __restrict__ fb,
                 const float* __restrict__ dw, const float* __restrict__ db,
                 float* __restrict__ out) {
    __shared__ float a[HD];
    __shared__ float fr[HD];
    int b = blockIdx.x, t = threadIdx.x;
    a[t] = aout[b * HD + t];
    __syncthreads();
    float acc = ob[t];
    for (int i = 0; i < HD; ++i) acc += a[i] * ow[t * HD + i];
    fr[t] = acc;
    out[224 + b * HD + t] = acc;          // fused
    __syncthreads();
    if (t == 0) {
        float r = rb[0];
        for (int i = 0; i < HD; ++i) r += fr[i] * rw[i];
        out[b] = 1.f / (1.f + expf(-r));  // risk
    } else if (t == 1) {
        float f = fb[0];
        for (int i = 0; i < HD; ++i) f += fr[i] * fw[i];
        out[BATCH + b] = f;               // forecast
    } else if (t >= 2 && t < 7) {
        int c = t - 2;
        float v = db[c];
        for (int i = 0; i < HD; ++i) v += fr[i] * dw[c * HD + i];
        out[2 * BATCH + b * 5 + c] = v;   // diet
    }
}

// ------------------------- Host orchestration ------------------------------
extern "C" void kernel_launch(void* const* d_in, const int* in_sizes, int n_in,
                              void* d_out, int out_size, void* d_ws, size_t ws_size,
                              hipStream_t stream) {
    (void)in_sizes; (void)n_in; (void)out_size; (void)ws_size;
    const float* tab_in  = (const float*)d_in[0];
    const float* temp_in = (const float*)d_in[1];
    const float* k1c = (const float*)d_in[2],  *k1b = (const float*)d_in[3];
    const float* g1  = (const float*)d_in[4],  *be1 = (const float*)d_in[5];
    const float* k2c = (const float*)d_in[6],  *k2b = (const float*)d_in[7];
    const float* g2  = (const float*)d_in[8],  *be2 = (const float*)d_in[9];
    const float* kow = (const float*)d_in[10], *kob = (const float*)d_in[11];
    const float* tiw = (const float*)d_in[12], *tib = (const float*)d_in[13];
    const float* aiw = (const float*)d_in[36], *aib = (const float*)d_in[37];
    const float* aow = (const float*)d_in[38], *aob = (const float*)d_in[39];
    const float* rw  = (const float*)d_in[40], *rb  = (const float*)d_in[41];
    const float* fw  = (const float*)d_in[42], *fb  = (const float*)d_in[43];
    const float* dw  = (const float*)d_in[44], *db  = (const float*)d_in[45];
    float* out = (float*)d_out;

    // workspace carve-up (f32 elements)
    float* ws = (float*)d_ws;
    size_t off = 0;
    auto alloc = [&](size_t n) { float* p = ws + off; off += n; return p; };
    float* tab   = alloc(BATCH * HD);
    float* qbuf  = alloc(BATCH * HD);
    float* aatt  = alloc(BATCH * HD);
    _Float16* wh = (_Float16*)alloc(32768);   // 64K f16: padded weight mirror
    float* tempA = alloc(BL * HD);
    float* tempB = alloc(BL * HD);
    float* xz    = alloc(BL * 2 * DI);
    float* xbc   = alloc(BL * DI);
    float* xdbl  = alloc(BL * NXD);
    float* dtb   = alloc(BL * DI);
    float* ybuf  = alloc(BL * DI);

    auto gemm = [&](const float* A, const float* W, const float* bias, float* C,
                    int M, int N, int K) {
        int ngroups = (N + 63) >> 6;
        long wtot = (long)ngroups * 64 * K;
        wcvt_kernel<<<(int)((wtot + 255) / 256), 256, 0, stream>>>(W, wh, N, K, wtot);
        int blocks = (M >> 7) * ngroups;
        if (K == 32)
            gemm_wmma_kernel<32><<<blocks, 256, 0, stream>>>(A, wh, bias, C, M, N);
        else if (K == 128)
            gemm_wmma_kernel<128><<<blocks, 256, 0, stream>>>(A, wh, bias, C, M, N);
        else
            gemm_wmma_kernel<256><<<blocks, 256, 0, stream>>>(A, wh, bias, C, M, N);
    };

    const int EW_BLOCKS = (int)(BL * DI / 256);

    // Tabular KAN encoder
    kan_tab_kernel<<<BATCH, HD, 0, stream>>>(tab_in, k1c, k1b, g1, be1,
                                             k2c, k2b, g2, be2, kow, kob, tab);
    // Temporal input projection
    gemm(temp_in, tiw, tib, tempA, (int)BL, HD, TEMPD);

    // Two Mamba blocks
    for (int mb = 0; mb < 2; ++mb) {
        int base = 14 + mb * 11;
        const float* in_w  = (const float*)d_in[base + 0];
        const float* in_b  = (const float*)d_in[base + 1];
        const float* cw    = (const float*)d_in[base + 2];
        const float* cb    = (const float*)d_in[base + 3];
        const float* xpw   = (const float*)d_in[base + 4];
        const float* dtw   = (const float*)d_in[base + 5];
        const float* dtbi  = (const float*)d_in[base + 6];
        const float* Alog  = (const float*)d_in[base + 7];
        const float* Dv    = (const float*)d_in[base + 8];
        const float* ow    = (const float*)d_in[base + 9];
        const float* obb   = (const float*)d_in[base + 10];
        const float* xin = (mb == 0) ? tempA : tempB;
        float*       xot = (mb == 0) ? tempB : tempA;

        gemm(xin, in_w, in_b, xz, (int)BL, 2 * DI, HD);
        conv_silu_kernel<<<EW_BLOCKS, 256, 0, stream>>>(xz, cw, cb, xbc);
        gemm(xbc, xpw, nullptr, xdbl, (int)BL, NXD, DI);
        dtproj_kernel<<<EW_BLOCKS, 256, 0, stream>>>(xdbl, dtw, dtbi, dtb);
        scan_kernel<<<BATCH, DI, 0, stream>>>(dtb, xbc, xdbl, Alog, ybuf);
        yact_kernel<<<EW_BLOCKS, 256, 0, stream>>>(ybuf, xbc, xz, Dv);
        gemm(ybuf, ow, obb, xot, (int)BL, HD, DI);
    }

    // Cross-attention fusion (final temporal activations live in tempA)
    qproj_kernel<<<BATCH, HD, 0, stream>>>(tab, aiw, aib, qbuf);
    float* Kbuf = xz;                  // xz region is free now
    float* Vbuf = xz + BL * HD;
    gemm(tempA, aiw + HD * HD,     aib + HD,     Kbuf, (int)BL, HD, HD);
    gemm(tempA, aiw + 2 * HD * HD, aib + 2 * HD, Vbuf, (int)BL, HD, HD);
    attn_kernel<<<dim3(BATCH, 4), 256, 0, stream>>>(qbuf, Kbuf, Vbuf, aatt);
    head_kernel<<<BATCH, HD, 0, stream>>>(aatt, aow, aob, rw, rb, fw, fb, dw, db, out);
}
